// VQBottleneck_60395830116472
// MI455X (gfx1250) — compile-verified, matrix operand order checked by reading, not run on hardware
//
#include <hip/hip_runtime.h>

// ---------------------------------------------------------------------------
// VQ cosine-codebook lookup for MI455X (gfx1250, wave32, WMMA).
//   dist = l2norm(x) @ l2norm(embed)^T   (f16 inputs, f32 accum, WMMA 16x16x32)
//   ind  = argmax_C dist                 (fused into GEMM, never materialized)
//   quantize = embed[ind]                (bandwidth-bound gather)
// ---------------------------------------------------------------------------

typedef _Float16 v16h __attribute__((ext_vector_type(16)));
typedef _Float16 v8h  __attribute__((ext_vector_type(8)));
typedef _Float16 v4h  __attribute__((ext_vector_type(4)));
typedef float    v8f  __attribute__((ext_vector_type(8)));
typedef float    v4f  __attribute__((ext_vector_type(4)));

#define BB 8
#define NN 4096
#define DD 512
#define CC 2048
#define ROWS (BB * NN)          // 32768 query rows
#define MTILE 128               // rows per workgroup (8 waves x 16 rows)
#define CTILES (CC / 16)        // 128 column tiles
#define KSTEPS (DD / 32)        // 16 WMMA K-steps per tile
// LDS B-tile row stride in halves: 512 + 24 pad -> 1072 bytes/row.
// 16B aligned (ds_*_b128 legal) and 1072/4 = 268 dwords, 268 % 64 = 12,
// 12*r mod 64 distinct for r=0..15 -> conflict-free 16-lane b128 reads.
#define LDS_STRIDE 536

// ---------------------------------------------------------------------------
// Kernel 1: renormalize codebook rows (matches F.normalize eps=1e-12) and
// convert to f16. One wave per row; lane t covers 4 strided float4 chunks.
// ---------------------------------------------------------------------------
__global__ __launch_bounds__(32) void cvt_embed_kernel(
    const float* __restrict__ embed, _Float16* __restrict__ ef16)
{
    const int row = blockIdx.x;
    const int t   = threadIdx.x;
    const float* src = embed + (size_t)row * DD;

    v4f f[4];
    float ss = 0.f;
#pragma unroll
    for (int g = 0; g < 4; ++g) {
        f[g] = *(const v4f*)(src + (t + 32 * g) * 4);
#pragma unroll
        for (int j = 0; j < 4; ++j) ss += f[g][j] * f[g][j];
    }
#pragma unroll
    for (int off = 16; off >= 1; off >>= 1) ss += __shfl_xor(ss, off, 32);
    const float scale = 1.0f / fmaxf(sqrtf(ss), 1e-12f);

    _Float16* dst = ef16 + (size_t)row * DD;
#pragma unroll
    for (int g = 0; g < 4; ++g) {
        v4h h;
#pragma unroll
        for (int j = 0; j < 4; ++j) h[j] = (_Float16)(f[g][j] * scale);
        *(v4h*)(dst + (t + 32 * g) * 4) = h;
    }
}

// ---------------------------------------------------------------------------
// Kernel 2: fused normalize(x) -> WMMA GEMM vs f16 codebook -> running argmax.
// 256 threads = 8 waves; wave w owns rows [blk*128 + w*16, +16).
// ---------------------------------------------------------------------------
__global__ __launch_bounds__(256, 1) void vq_argmax_kernel(
    const float* __restrict__ x, const _Float16* __restrict__ ef16,
    int* __restrict__ idxbuf, float* __restrict__ ind_out)
{
    __shared__ _Float16 btile[2][16 * LDS_STRIDE];

    const int tid  = threadIdx.x;
    const int w    = tid >> 5;
    const int l    = tid & 31;
    const int lrow = l & 15;        // A: row-in-16; B: col-in-16; D: column
    const int half = l >> 4;
    const int hb   = half * 8;      // K sub-offset per ISA 16-bit A/B layout

    const int rowBase  = blockIdx.x * MTILE + w * 16;
    const float* xrow  = x + (size_t)(rowBase + lrow) * DD;

    // ---- pass 1: row norm. Lane pair (l, l+16) splits K in halves of 256.
    float ss = 0.f;
    {
        const v4f* p = (const v4f*)(xrow + half * 256);
#pragma unroll 8
        for (int i = 0; i < 64; ++i) {
            v4f f = p[i];
            ss += f[0] * f[0] + f[1] * f[1] + f[2] * f[2] + f[3] * f[3];
        }
    }
    ss += __shfl_xor(ss, 16, 32);
    const float scale = 1.0f / fmaxf(sqrtf(ss), 1e-12f);

    // ---- pass 2: resident A fragments, full K=512 (16 x v16h = 128 VGPRs).
    // Lane l, frag kk: halves 0..7 -> K = kk*32 + hb + j ; 8..15 -> +16.
    v16h a[KSTEPS];
#pragma unroll
    for (int kk = 0; kk < KSTEPS; ++kk) {
        const float* pb = xrow + kk * 32 + hb;
        v4f f0 = *(const v4f*)(pb);
        v4f f1 = *(const v4f*)(pb + 4);
        v4f f2 = *(const v4f*)(pb + 16);
        v4f f3 = *(const v4f*)(pb + 20);
        v16h af;
#pragma unroll
        for (int j = 0; j < 4; ++j) {
            af[j]      = (_Float16)(f0[j] * scale);
            af[4 + j]  = (_Float16)(f1[j] * scale);
            af[8 + j]  = (_Float16)(f2[j] * scale);
            af[12 + j] = (_Float16)(f3[j] * scale);
        }
        a[kk] = af;
    }

    // ---- LDS staging plan: tile = 16 embed rows x 512 halves = 1024 x 16B.
    // Thread tid copies 4 chunks (tid + i*256).
    const v8h* src0 = (const v8h*)ef16;
    int ldsoff[4];
#pragma unroll
    for (int i = 0; i < 4; ++i) {
        const int c = tid + i * 256;
        ldsoff[i] = (c >> 6) * LDS_STRIDE + (c & 63) * 8;
    }
    // prologue: stage tile 0
#pragma unroll
    for (int i = 0; i < 4; ++i)
        *(v8h*)&btile[0][ldsoff[i]] = src0[tid + i * 256];

    float bestv[8];
    int   besti[8];
#pragma unroll
    for (int v = 0; v < 8; ++v) { bestv[v] = -3.4e38f; besti[v] = 0; }

    for (int ct = 0; ct < CTILES; ++ct) {
        __syncthreads();

        // register-staged prefetch of next tile (overlaps with WMMAs below)
        const bool more = (ct + 1 < CTILES);
        v8h g[4];
        if (more) {
            const v8h* s = src0 + (size_t)(ct + 1) * 1024;
#pragma unroll
            for (int i = 0; i < 4; ++i) g[i] = s[tid + i * 256];
        }

        // 16x16 tile: C columns [ct*16, ct*16+16), full K via 16 WMMAs
        const _Float16* bt = &btile[ct & 1][lrow * LDS_STRIDE];
        v8f acc = {0.f, 0.f, 0.f, 0.f, 0.f, 0.f, 0.f, 0.f};
#pragma unroll
        for (int kk = 0; kk < KSTEPS; ++kk) {
            v8h lo = *(const v8h*)(bt + kk * 32 + hb);
            v8h hi = *(const v8h*)(bt + kk * 32 + 16 + hb);
            v16h bf = __builtin_shufflevector(lo, hi,
                0, 1, 2, 3, 4, 5, 6, 7, 8, 9, 10, 11, 12, 13, 14, 15);
            acc = __builtin_amdgcn_wmma_f32_16x16x32_f16(
                false, a[kk], false, bf, (short)0, acc, false, false);
        }

        if (more) {
#pragma unroll
            for (int i = 0; i < 4; ++i)
                *(v8h*)&btile[(ct + 1) & 1][ldsoff[i]] = g[i];
        }

        // running argmax; columns scan upward so strict '>' keeps first winner
        const int col = ct * 16 + lrow;
#pragma unroll
        for (int v = 0; v < 8; ++v) {
            const float av = acc[v];
            if (av > bestv[v]) { bestv[v] = av; besti[v] = col; }
        }
    }

    // ---- per-row reduction across the 16 lanes of each half
    // slot v in half h corresponds to row v + 8*h (ISA C/D layout)
#pragma unroll
    for (int off = 1; off < 16; off <<= 1) {
#pragma unroll
        for (int v = 0; v < 8; ++v) {
            const float ov = __shfl_xor(bestv[v], off, 32);
            const int   oi = __shfl_xor(besti[v], off, 32);
            if (ov > bestv[v] || (ov == bestv[v] && oi < besti[v])) {
                bestv[v] = ov; besti[v] = oi;
            }
        }
    }
    if (lrow == 0) {
#pragma unroll
        for (int v = 0; v < 8; ++v) {
            const int grow = rowBase + v + 8 * half;
            idxbuf[grow]  = besti[v];
            ind_out[grow] = (float)besti[v];
        }
    }
}

// ---------------------------------------------------------------------------
// Kernel 3: quantize = embed[ind]. One block per row; float4 copies.
// embed (4 MB) is L2-resident, so this runs at the HBM store limit.
// ---------------------------------------------------------------------------
__global__ __launch_bounds__(128) void gather_kernel(
    const float* __restrict__ embed, const int* __restrict__ idxbuf,
    float* __restrict__ outq)
{
    const int row = blockIdx.x;
    const int idx = idxbuf[row];
    const v4f* src = (const v4f*)(embed + (size_t)idx * DD);
    v4f*       dst = (v4f*)(outq + (size_t)row * DD);
    dst[threadIdx.x] = src[threadIdx.x];
}

// ---------------------------------------------------------------------------
extern "C" void kernel_launch(void* const* d_in, const int* in_sizes, int n_in,
                              void* d_out, int out_size, void* d_ws, size_t ws_size,
                              hipStream_t stream)
{
    const float* x     = (const float*)d_in[0];   // [8,4096,512] f32
    const float* embed = (const float*)d_in[1];   // [1,2048,512] f32 (unit rows)

    float* out     = (float*)d_out;
    float* outq    = out;                          // [8,4096,512]
    float* outind  = out + (size_t)ROWS * DD;      // [8,4096] (as float)

    _Float16* ef16 = (_Float16*)d_ws;                                  // 2 MB
    int* idxbuf    = (int*)((char*)d_ws + (size_t)CC * DD * sizeof(_Float16));

    cvt_embed_kernel<<<CC, 32, 0, stream>>>(embed, ef16);
    vq_argmax_kernel<<<ROWS / MTILE, 256, 0, stream>>>(x, ef16, idxbuf, outind);
    gather_kernel<<<ROWS, 128, 0, stream>>>(embed, idxbuf, outq);
}
